// SelfAttention_61684320305475
// MI455X (gfx1250) — compile-verified, matrix operand order checked by reading, not run on hardware
//
#include <hip/hip_runtime.h>
#include <stdint.h>

typedef __attribute__((ext_vector_type(16))) _Float16 v16h;
typedef __attribute__((ext_vector_type(8)))  _Float16 v8h;
typedef __attribute__((ext_vector_type(8)))  float    v8f;
typedef __attribute__((ext_vector_type(4)))  float    v4f;

#define SEQ       8192
#define DM        512
#define DK        64
#define INV_SCALE 0.125f
#define QTILES    8                 // query tiles (waves) per attention block
#define KPAD      72                // LDS row stride for K slab (64+8 halves)
#define VPAD      40                // LDS row stride for V slab (32+8 halves)
#define KSLAB     (32 * KPAD)       // halves
#define VSLAB     (64 * VPAD)       // halves

// Load 8+8 contiguous f32 and convert to a v16h WMMA operand.
static __device__ inline v16h cvt16(const float* __restrict__ p0,
                                    const float* __restrict__ p1) {
  v4f a0 = *(const v4f*)p0;
  v4f a1 = *(const v4f*)(p0 + 4);
  v4f b0 = *(const v4f*)p1;
  v4f b1 = *(const v4f*)(p1 + 4);
  v16h r;
#pragma unroll
  for (int i = 0; i < 4; ++i) {
    r[i]      = (_Float16)a0[i];
    r[i + 4]  = (_Float16)a1[i];
    r[i + 8]  = (_Float16)b0[i];
    r[i + 12] = (_Float16)b1[i];
  }
  return r;
}

// Load 8+8 contiguous f16 (two b128 loads, global or LDS) into a v16h operand.
static __device__ inline v16h ldh16(const _Float16* p0, const _Float16* p1) {
  v8h lo = *(const v8h*)p0;
  v8h hi = *(const v8h*)p1;
  v16h r;
#pragma unroll
  for (int i = 0; i < 8; ++i) { r[i] = lo[i]; r[i + 8] = hi[i]; }
  return r;
}

// gfx1250 async DMA: global -> LDS, 16 bytes, tracked by ASYNCcnt.
static __device__ inline void async_b128(uint32_t lds_off, const void* gptr) {
  asm volatile("global_load_async_to_lds_b128 %0, %1, off"
               :: "v"(lds_off), "v"((uint64_t)(uintptr_t)gptr) : "memory");
}
static __device__ inline void wait_async0() {
  asm volatile("s_wait_asynccnt 0" ::: "memory");
}

// ---------------------------------------------------------------------------
// Kernel 0: f32 -> f16 weights, Wh = [3][64][512].
// ---------------------------------------------------------------------------
__global__ void __launch_bounds__(256)
wcvt_kernel(const float* __restrict__ Wq, const float* __restrict__ Wk,
            const float* __restrict__ Wv, _Float16* __restrict__ Wh) {
  const int i = blockIdx.x * blockDim.x + threadIdx.x;
  const int n = DK * DM;
  const float* src = (i < n) ? Wq : ((i < 2 * n) ? Wk : Wv);
  Wh[i] = (_Float16)src[i % n];
}

// ---------------------------------------------------------------------------
// Kernel 1: projections. One wave computes ONE matrix (Q, K or V) for a
// 16-row tile (32 accumulator VGPRs -> good load pipelining).
// ---------------------------------------------------------------------------
__global__ void __launch_bounds__(128)
proj_kernel(const float* __restrict__ x, const _Float16* __restrict__ Wh,
            const float* __restrict__ bq, const float* __restrict__ bk,
            const float* __restrict__ bv,
            _Float16* __restrict__ Qh, _Float16* __restrict__ Kh,
            _Float16* __restrict__ Vt) {
  const int gw   = blockIdx.x * (blockDim.x >> 5) + (threadIdx.x >> 5);
  const int lane = threadIdx.x & 31;
  const int l15  = lane & 15;
  const int hb   = (lane >> 4) << 3;
  const int mat  = gw % 3;
  const int s0   = (gw / 3) * 16;

  const _Float16* W = Wh + (size_t)mat * DK * DM;
  const float*  xrow = x + (size_t)(s0 + l15) * DM;

  v8f acc[4];
#pragma unroll
  for (int t = 0; t < 4; ++t)
#pragma unroll
    for (int i = 0; i < 8; ++i) acc[t][i] = 0.f;

  if (mat == 2) {                        // V: C = x @ Wv^T -> Vt[64][SEQ]
    for (int kk = 0; kk < DM; kk += 32) {
      v16h xin = cvt16(xrow + kk + hb, xrow + kk + 16 + hb);
#pragma unroll
      for (int db = 0; db < 4; ++db) {
        const _Float16* wr = W + (size_t)(db * 16 + l15) * DM + kk;
        v16h w = ldh16(wr + hb, wr + 16 + hb);
        acc[db] = __builtin_amdgcn_wmma_f32_16x16x32_f16(false, xin, false, w,
                                                         (short)0, acc[db], false, false);
      }
    }
#pragma unroll
    for (int db = 0; db < 4; ++db) {
      const int dv = db * 16 + l15;
      const float b = bv[dv];
      v8h ov;
#pragma unroll
      for (int i = 0; i < 8; ++i) ov[i] = (_Float16)(acc[db][i] + b);
      *(v8h*)(Vt + (size_t)dv * SEQ + s0 + hb) = ov;
    }
  } else {                               // Q/K: C = W @ x^T -> row-major
    for (int kk = 0; kk < DM; kk += 32) {
      v16h xin = cvt16(xrow + kk + hb, xrow + kk + 16 + hb);
#pragma unroll
      for (int db = 0; db < 4; ++db) {
        const _Float16* wr = W + (size_t)(db * 16 + l15) * DM + kk;
        v16h w = ldh16(wr + hb, wr + 16 + hb);
        acc[db] = __builtin_amdgcn_wmma_f32_16x16x32_f16(false, w, false, xin,
                                                         (short)0, acc[db], false, false);
      }
    }
    const float* bias = (mat == 0) ? bq : bk;
    const float  scl  = (mat == 0) ? INV_SCALE : 1.0f;
    _Float16*    dst  = (mat == 0) ? Qh : Kh;
#pragma unroll
    for (int db = 0; db < 4; ++db) {
      const int d0 = db * 16 + hb;
      v8h o;
#pragma unroll
      for (int i = 0; i < 8; ++i)
        o[i] = (_Float16)((acc[db][i] + bias[d0 + i]) * scl);
      *(v8h*)(dst + (size_t)(s0 + l15) * DK + d0) = o;
    }
  }
}

// ---------------------------------------------------------------------------
// Kernel 2: flash attention, LDS-staged K/V shared by 8 waves.
//   block = 8 waves = 8 query tiles (128 rows); blockIdx.y = key split.
//   Per 32-key step: async-DMA the next K[32][64]/Vt[64][32] slab into LDS
//   (double buffered) while computing on the current one.
//   S^T = K_blk @ Q^T ; O^T += V^T @ P^T  (P operand == exp(S^T) registers).
//   Writes unnormalized partial O^T plus (m, l) per row per split.
// ---------------------------------------------------------------------------
__global__ void __launch_bounds__(256)
flash_attn_kernel(const _Float16* __restrict__ Qh,
                  const _Float16* __restrict__ Kh,
                  const _Float16* __restrict__ Vt,
                  float* __restrict__ Pm, float* __restrict__ Pl,
                  float* __restrict__ Pacc, int nkeys, int nit) {
  __shared__ _Float16 smem[2][KSLAB + VSLAB];

  const int tid  = threadIdx.x;
  const int widx = tid >> 5;
  const int lane = tid & 31;
  const int l15  = lane & 15;
  const int hb   = (lane >> 4) << 3;
  const int kz   = blockIdx.y;
  const int s0   = (blockIdx.x * QTILES + widx) * 16;
  const int mrow = s0 + l15;
  const int kb0  = kz * nkeys;

  // staging addresses for this thread (one b128 of K, one b128 of V per slab)
  const int krow = tid >> 3, kcol = (tid & 7) << 3;   // 32 rows x 64 halves
  const int vrow = tid >> 2, vcol = (tid & 3) << 3;   // 64 rows x 32 halves

  // Q row of this lane -> two B operands (d 0-31, d 32-63)
  const _Float16* qrow = Qh + (size_t)mrow * DK;
  const v16h qB01 = ldh16(qrow + hb,      qrow + 16 + hb);
  const v16h qB23 = ldh16(qrow + 32 + hb, qrow + 48 + hb);

  v8f acc[4];
#pragma unroll
  for (int t = 0; t < 4; ++t)
#pragma unroll
    for (int i = 0; i < 8; ++i) acc[t][i] = 0.f;

  float mi = -__builtin_huge_valf();
  float li = 0.f;

  // prologue: stage slab 0
  async_b128((uint32_t)(uintptr_t)(&smem[0][0] + krow * KPAD + kcol),
             Kh + (size_t)(kb0 + krow) * DK + kcol);
  async_b128((uint32_t)(uintptr_t)(&smem[0][KSLAB] + vrow * VPAD + vcol),
             Vt + (size_t)vrow * SEQ + kb0 + vcol);
  wait_async0();
  __syncthreads();

  int kb = kb0;
  for (int it = 0; it < nit; ++it) {
    const int cur = it & 1;
    const _Float16* ldsK = &smem[cur][0];
    const _Float16* ldsV = &smem[cur][KSLAB];

    // ---- kick off async DMA of the next slab into the other buffer ----
    if (it + 1 < nit) {
      const int kn = kb + 32;
      async_b128((uint32_t)(uintptr_t)(&smem[cur ^ 1][0] + krow * KPAD + kcol),
                 Kh + (size_t)(kn + krow) * DK + kcol);
      async_b128((uint32_t)(uintptr_t)(&smem[cur ^ 1][KSLAB] + vrow * VPAD + vcol),
                 Vt + (size_t)vrow * SEQ + kn + vcol);
    }

    // ---- S^T from LDS-resident K slab ----
    v8f st[2];
#pragma unroll
    for (int t = 0; t < 2; ++t) {
      const _Float16* kr = ldsK + (16 * t + l15) * KPAD;
      v16h ka0 = ldh16(kr + hb,      kr + 16 + hb);
      v16h ka1 = ldh16(kr + 32 + hb, kr + 48 + hb);
      v8f c;
#pragma unroll
      for (int i = 0; i < 8; ++i) c[i] = 0.f;
      c = __builtin_amdgcn_wmma_f32_16x16x32_f16(false, ka0, false, qB01,
                                                 (short)0, c, false, false);
      c = __builtin_amdgcn_wmma_f32_16x16x32_f16(false, ka1, false, qB23,
                                                 (short)0, c, false, false);
      st[t] = c;
    }

    // ---- online softmax over the 32-key block ----
    float vmax = st[0][0];
#pragma unroll
    for (int i = 0; i < 8; ++i) {
      vmax = fmaxf(vmax, st[0][i]);
      vmax = fmaxf(vmax, st[1][i]);
    }
    vmax = fmaxf(vmax, __shfl_xor(vmax, 16, 32));
    const float mnew  = fmaxf(mi, vmax);
    const float alpha = __expf(mi - mnew);

    float rsum = 0.f;
    v16h pa;
#pragma unroll
    for (int i = 0; i < 8; ++i) {
      float p0 = __expf(st[0][i] - mnew);
      float p1 = __expf(st[1][i] - mnew);
      rsum += p0 + p1;
      pa[i]     = (_Float16)p0;
      pa[i + 8] = (_Float16)p1;
    }
    rsum += __shfl_xor(rsum, 16, 32);
    li = li * alpha + rsum;
    mi = mnew;

#pragma unroll
    for (int db = 0; db < 4; ++db)
#pragma unroll
      for (int i = 0; i < 8; ++i) acc[db][i] *= alpha;

    // ---- O^T += V^T @ P^T from LDS-resident V slab ----
#pragma unroll
    for (int db = 0; db < 4; ++db) {
      const _Float16* vr = ldsV + (db * 16 + l15) * VPAD;
      v16h va = ldh16(vr + hb, vr + 16 + hb);
      acc[db] = __builtin_amdgcn_wmma_f32_16x16x32_f16(false, va, false, pa,
                                                       (short)0, acc[db], false, false);
    }

    if (it + 1 < nit) wait_async0();   // next slab landed
    __syncthreads();                   // everyone done with cur, next visible
    kb += 32;
  }

  // ---- write unnormalized partials ----
  if (lane < 16) {
    Pm[(size_t)kz * SEQ + mrow] = mi;
    Pl[(size_t)kz * SEQ + mrow] = li;
  }
  float* prow = Pacc + ((size_t)kz * SEQ + mrow) * DK;
#pragma unroll
  for (int db = 0; db < 4; ++db) {
    v4f o0, o1;
#pragma unroll
    for (int i = 0; i < 4; ++i) { o0[i] = acc[db][i]; o1[i] = acc[db][i + 4]; }
    *(v4f*)(prow + db * 16 + hb)     = o0;
    *(v4f*)(prow + db * 16 + hb + 4) = o1;
  }
}

// ---------------------------------------------------------------------------
// Kernel 3: cross-split log-sum-exp merge + normalization.
// One thread per (row, 16-wide d-chunk).
// ---------------------------------------------------------------------------
__global__ void __launch_bounds__(256)
merge_kernel(const float* __restrict__ Pm, const float* __restrict__ Pl,
             const float* __restrict__ Pacc, float* __restrict__ out, int S) {
  const int tid = blockIdx.x * blockDim.x + threadIdx.x;   // SEQ*4 threads
  const int r  = tid >> 2;
  const int d0 = (tid & 3) << 4;

  float m[4];
  float mtot = -__builtin_huge_valf();
  for (int s = 0; s < S; ++s) { m[s] = Pm[(size_t)s * SEQ + r]; mtot = fmaxf(mtot, m[s]); }

  float ltot = 0.f;
  v4f o[4];
#pragma unroll
  for (int j = 0; j < 4; ++j)
#pragma unroll
    for (int i = 0; i < 4; ++i) o[j][i] = 0.f;

  for (int s = 0; s < S; ++s) {
    const float f = __expf(m[s] - mtot);
    ltot += f * Pl[(size_t)s * SEQ + r];
    const float* pa = Pacc + ((size_t)s * SEQ + r) * DK + d0;
#pragma unroll
    for (int j = 0; j < 4; ++j) {
      v4f t = *(const v4f*)(pa + j * 4);
      o[j] += t * f;
    }
  }
  const float inv_l = 1.f / ltot;
  float* orow = out + (size_t)r * DK + d0;
#pragma unroll
  for (int j = 0; j < 4; ++j) *(v4f*)(orow + j * 4) = o[j] * inv_l;
}

extern "C" void kernel_launch(void* const* d_in, const int* in_sizes, int n_in,
                              void* d_out, int out_size, void* d_ws, size_t ws_size,
                              hipStream_t stream) {
  const float* x  = (const float*)d_in[0];
  const float* Wq = (const float*)d_in[1];
  const float* bq = (const float*)d_in[2];
  const float* Wk = (const float*)d_in[3];
  const float* bk = (const float*)d_in[4];
  const float* Wv = (const float*)d_in[5];
  const float* bv = (const float*)d_in[6];
  float* out = (float*)d_out;

  // ws: Qh | Kh | Vt | Wh (f16)  then  Pm | Pl | Pacc (f32) per key-split
  _Float16* Qh = (_Float16*)d_ws;
  _Float16* Kh = Qh + (size_t)SEQ * DK;
  _Float16* Vt = Kh + (size_t)SEQ * DK;
  _Float16* Wh = Vt + (size_t)SEQ * DK;
  const size_t fixed = (size_t)SEQ * DK * 2 * 3 + (size_t)3 * DK * DM * 2;

  int KS = 4;                                   // key splits across blocks
  while (KS > 1 &&
         fixed + (size_t)KS * ((size_t)SEQ * 8 + (size_t)SEQ * DK * 4) > ws_size)
    KS >>= 1;

  float* Pm   = (float*)((char*)d_ws + fixed);
  float* Pl   = Pm + (size_t)KS * SEQ;
  float* Pacc = Pl + (size_t)KS * SEQ;

  wcvt_kernel<<<dim3(3 * DK * DM / 256), dim3(256), 0, stream>>>(Wq, Wk, Wv, Wh);

  proj_kernel<<<dim3(1536 / 4), dim3(128), 0, stream>>>(x, Wh, bq, bk, bv, Qh, Kh, Vt);

  const int nkeys = SEQ / KS;
  flash_attn_kernel<<<dim3(SEQ / 16 / QTILES, KS), dim3(256), 0, stream>>>(
      Qh, Kh, Vt, Pm, Pl, Pacc, nkeys, nkeys / 32);

  merge_kernel<<<dim3(SEQ * 4 / 256), dim3(256), 0, stream>>>(Pm, Pl, Pacc, out, KS);
}
